// DynamicMemoryBank_39565238731324
// MI455X (gfx1250) — compile-verified, matrix operand order checked by reading, not run on hardware
//
#include <hip/hip_runtime.h>

#define B_Q     1024
#define D_DIM   512
#define M_ROWS  262144
#define F_DIM   256
#define TOPK    8
#define NCHUNK  256
#define CHUNK_ROWS 1024
#define SUBS    8
#define SUBROWS 128
#define QTILES  64

#define OFF_FLOW  4194304u   // 1024*8*512
#define OFF_W     6291456u   // + 1024*8*256

typedef __attribute__((ext_vector_type(16))) __bf16 v16bf;
typedef __attribute__((ext_vector_type(8)))  float  v8f;
typedef __attribute__((ext_vector_type(4)))  unsigned int v4u;
typedef __attribute__((ext_vector_type(2)))  unsigned int v2u;

union FragBF { v4u u[2]; v16bf f; };

__device__ __forceinline__ unsigned short f2bf(float x) {
  union { float f; unsigned int u; } c; c.f = x;
  unsigned int r = c.u + 0x7FFFu + ((c.u >> 16) & 1u);   // round-to-nearest-even
  return (unsigned short)(r >> 16);
}

__device__ __forceinline__ float wave_reduce_sum(float s) {
#pragma unroll
  for (int off = 16; off > 0; off >>= 1)
    s += __shfl_xor(s, off, 32);
  return s;
}

// ---------------- query normalize -> bf16 ----------------
__global__ __launch_bounds__(256) void qnorm_kernel(const float* __restrict__ q,
                                                    unsigned short* __restrict__ qbf) {
  const int wave = threadIdx.x >> 5, lane = threadIdx.x & 31;
  const int row = blockIdx.x * 8 + wave;
  const float4* src = (const float4*)(q + (size_t)row * D_DIM);
  float4 v[4]; float s = 0.f;
#pragma unroll
  for (int j = 0; j < 4; ++j) {
    v[j] = src[lane + j * 32];
    s += v[j].x * v[j].x + v[j].y * v[j].y + v[j].z * v[j].z + v[j].w * v[j].w;
  }
  s = wave_reduce_sum(s);
  const float scale = 1.0f / fmaxf(sqrtf(s), 1e-12f);
  unsigned short* dst = qbf + (size_t)row * D_DIM;
#pragma unroll
  for (int j = 0; j < 4; ++j) {
    const int col = (lane + j * 32) * 4;
    v2u p;
    p[0] = (unsigned)f2bf(v[j].x * scale) | ((unsigned)f2bf(v[j].y * scale) << 16);
    p[1] = (unsigned)f2bf(v[j].z * scale) | ((unsigned)f2bf(v[j].w * scale) << 16);
    *(v2u*)(dst + col) = p;
  }
}

// ---------------- memory row inverse norms ----------------
__global__ __launch_bounds__(256) void mnorm_kernel(const float* __restrict__ m,
                                                    float* __restrict__ invm) {
  const int wave = threadIdx.x >> 5, lane = threadIdx.x & 31;
  const size_t row = (size_t)blockIdx.x * 8 + wave;
  const float4* src = (const float4*)(m + row * D_DIM);
  float s = 0.f;
#pragma unroll
  for (int j = 0; j < 4; ++j) {
    float4 v = src[lane + j * 32];
    s += v.x * v.x + v.y * v.y + v.z * v.z + v.w * v.w;
  }
  s = wave_reduce_sum(s);
  if (lane == 0) invm[row] = 1.0f / fmaxf(sqrtf(s), 1e-12f);
}

// ------- fused GEMM (bf16 WMMA) + per-chunk top-8 -------
__global__ __launch_bounds__(256) void sims_topk_kernel(const float* __restrict__ memf,
                                                        const unsigned short* __restrict__ qbf,
                                                        const float* __restrict__ invm,
                                                        float* __restrict__ cand_val,
                                                        int* __restrict__ cand_idx) {
  __shared__ __align__(16) unsigned short memLDS[SUBROWS * D_DIM]; // 128 KB bf16 tile
  __shared__ __align__(16) float simsLDS[16 * SUBROWS];            // 8 KB
  __shared__ float invLDS[SUBROWS];

  const int tid = threadIdx.x;
  const int chunk = blockIdx.x;
  const long mbase = (long)chunk * CHUNK_ROWS;

  // init this chunk's candidate slots for all queries (deterministic each launch)
  for (int i = tid; i < B_Q * TOPK; i += 256) {
    const int b = i >> 3, j = i & 7;
    const size_t o = (size_t)b * (NCHUNK * TOPK) + (size_t)chunk * TOPK + j;
    cand_val[o] = -3.0e38f;
    cand_idx[o] = 0;
  }
  __threadfence_block();

  const int wave = tid >> 5;
  const int lane = tid & 31;
  const int lrow = lane & 15;
  const int ks   = (lane >> 4) << 3;       // 0 or 8 per ISA 16-bit A/B layout

  for (int s = 0; s < SUBS; ++s) {
    const long rowbase = mbase + (long)s * SUBROWS;
    __syncthreads();                        // memLDS no longer being read
    if (tid < SUBROWS) invLDS[tid] = invm[rowbase + tid];
    // stage 128x512 f32 tile -> bf16 in LDS (read once from HBM)
    for (int i = tid; i < SUBROWS * D_DIM / 4; i += 256) {
      const int e = i << 2;
      const int r = e >> 9;
      const int k = e & (D_DIM - 1);
      float4 v = *(const float4*)(memf + ((size_t)(rowbase + r) << 9) + k);
      v2u p;
      p[0] = (unsigned)f2bf(v.x) | ((unsigned)f2bf(v.y) << 16);
      p[1] = (unsigned)f2bf(v.z) | ((unsigned)f2bf(v.w) << 16);
      *(v2u*)(&memLDS[r * D_DIM + k]) = p;
    }
    __syncthreads();

    const unsigned short* brow = &memLDS[(wave * 16 + lrow) * D_DIM];
    const float invc = invLDS[wave * 16 + lrow];

    for (int qt = 0; qt < QTILES; ++qt) {
      const unsigned short* arow = qbf + (size_t)(qt * 16 + lrow) * D_DIM;
      v8f acc = {0.f, 0.f, 0.f, 0.f, 0.f, 0.f, 0.f, 0.f};
#pragma unroll
      for (int kb = 0; kb < D_DIM / 32; ++kb) {
        const int k0 = kb * 32 + ks;
        FragBF A, Bm;
        A.u[0]  = *(const v4u*)(arow + k0);
        A.u[1]  = *(const v4u*)(arow + k0 + 16);
        Bm.u[0] = *(const v4u*)(brow + k0);
        Bm.u[1] = *(const v4u*)(brow + k0 + 16);
        acc = __builtin_amdgcn_wmma_f32_16x16x32_bf16(false, A.f, false, Bm.f,
                                                      (short)0, acc, false, false);
      }
      __syncthreads();   // previous qt's top-8 scan of simsLDS finished
      {
        const int col = wave * 16 + lrow;
#pragma unroll
        for (int r = 0; r < 8; ++r) {
          const int qq = r + ((lane >> 4) << 3);   // C layout: M = r + 8*(lane>=16)
          simsLDS[qq * SUBROWS + col] = acc[r] * invc;
        }
      }
      __syncthreads();
      if (tid < 16) {   // one thread per query: update running chunk top-8
        const int b = qt * 16 + tid;
        const size_t base = (size_t)b * (NCHUNK * TOPK) + (size_t)chunk * TOPK;
        float tv[8]; int ti[8];
#pragma unroll
        for (int j = 0; j < 8; ++j) { tv[j] = cand_val[base + j]; ti[j] = cand_idx[base + j]; }
        const float* srow = &simsLDS[tid * SUBROWS];
        for (int col = 0; col < SUBROWS; ++col) {
          const float v = srow[col];
          if (v > tv[7]) {
            tv[7] = v; ti[7] = (int)(rowbase + col);
#pragma unroll
            for (int j = 7; j > 0; --j) {
              if (tv[j] > tv[j - 1]) {
                float fv = tv[j]; tv[j] = tv[j - 1]; tv[j - 1] = fv;
                int   iv = ti[j]; ti[j] = ti[j - 1]; ti[j - 1] = iv;
              }
            }
          }
        }
#pragma unroll
        for (int j = 0; j < 8; ++j) { cand_val[base + j] = tv[j]; cand_idx[base + j] = ti[j]; }
      }
    }
  }
}

// ---------------- merge candidates + softmax ----------------
__global__ __launch_bounds__(256) void merge_kernel(const float* __restrict__ cand_val,
                                                    const int* __restrict__ cand_idx,
                                                    float* __restrict__ out_w,
                                                    int* __restrict__ fidx) {
  const int b = blockIdx.x * 256 + threadIdx.x;
  if (b >= B_Q) return;
  const float* cv = cand_val + (size_t)b * (NCHUNK * TOPK);
  const int*   ci = cand_idx + (size_t)b * (NCHUNK * TOPK);
  float tv[8]; int ti[8];
#pragma unroll
  for (int j = 0; j < 8; ++j) { tv[j] = -3.0e38f; ti[j] = 0; }
  for (int i = 0; i < NCHUNK * TOPK; ++i) {
    const float v = cv[i];
    if (v > tv[7]) {
      tv[7] = v; ti[7] = ci[i];
#pragma unroll
      for (int j = 7; j > 0; --j) {
        if (tv[j] > tv[j - 1]) {
          float fv = tv[j]; tv[j] = tv[j - 1]; tv[j - 1] = fv;
          int   iv = ti[j]; ti[j] = ti[j - 1]; ti[j - 1] = iv;
        }
      }
    }
  }
  const float mx = tv[0];
  float e[8]; float sum = 0.f;
#pragma unroll
  for (int j = 0; j < 8; ++j) { e[j] = __expf((tv[j] - mx) * 10.0f); sum += e[j]; }
  const float rs = 1.0f / sum;
#pragma unroll
  for (int j = 0; j < 8; ++j) {
    out_w[(size_t)b * 8 + j] = e[j] * rs;
    fidx[(size_t)b * 8 + j]  = ti[j];
  }
}

// ---------------- gather features + flows ----------------
__global__ __launch_bounds__(256) void gather_kernel(const float* __restrict__ memf,
                                                     const float* __restrict__ flows,
                                                     const int* __restrict__ fidx,
                                                     float* __restrict__ out) {
  const int bk = blockIdx.x;            // b*8 + k
  const int idx = fidx[bk];
  const int t = threadIdx.x;
  if (t < 128) {
    const float4* s = (const float4*)(memf + (size_t)idx * D_DIM);
    float4* d = (float4*)(out + (size_t)bk * D_DIM);
    d[t] = s[t];
  } else if (t < 192) {
    const float4* s = (const float4*)(flows + (size_t)idx * F_DIM);
    float4* d = (float4*)(out + OFF_FLOW + (size_t)bk * F_DIM);
    d[t - 128] = s[t - 128];
  }
}

extern "C" void kernel_launch(void* const* d_in, const int* in_sizes, int n_in,
                              void* d_out, int out_size, void* d_ws, size_t ws_size,
                              hipStream_t stream) {
  (void)in_sizes; (void)n_in; (void)out_size; (void)ws_size;
  const float* q     = (const float*)d_in[0];
  const float* memf  = (const float*)d_in[1];
  const float* flows = (const float*)d_in[2];
  // d_in[3] = top_k scalar (always 8 for this problem)

  char* ws = (char*)d_ws;
  unsigned short* qbf = (unsigned short*)ws;                 // 1 MB  bf16 normalized queries
  float* invm     = (float*)(ws + (1u << 20));               // 1 MB  1/||m||
  float* cand_val = (float*)(ws + (2u << 20));               // 8 MB  [B][NCHUNK][8]
  int*   cand_idx = (int*)  (ws + (10u << 20));              // 8 MB
  int*   fidx     = (int*)  (ws + (18u << 20));              // 32 KB final indices
  float* out = (float*)d_out;

  qnorm_kernel<<<B_Q / 8, 256, 0, stream>>>(q, qbf);
  mnorm_kernel<<<M_ROWS / 8, 256, 0, stream>>>(memf, invm);
  sims_topk_kernel<<<NCHUNK, 256, 0, stream>>>(memf, qbf, invm, cand_val, cand_idx);
  merge_kernel<<<(B_Q + 255) / 256, 256, 0, stream>>>(cand_val, cand_idx, out + OFF_W, fidx);
  gather_kernel<<<B_Q * TOPK, 256, 0, stream>>>(memf, flows, fidx, out);
}
